// MultiHeadedAttention_4269197492266
// MI455X (gfx1250) — matrix operand order repaired and checked by reading, not compiled, not run on hardware
//
#include <hip/hip_runtime.h>
#include <hip/hip_bf16.h>
#include <math.h>

typedef __bf16 bf16;
typedef __attribute__((ext_vector_type(16))) __bf16 v16bf;
typedef __attribute__((ext_vector_type(8)))  __bf16 v8bf;
typedef __attribute__((ext_vector_type(8)))  float  v8f;
typedef __attribute__((ext_vector_type(4)))  unsigned int u32x4;
typedef __attribute__((ext_vector_type(8)))  int i32x8;
typedef __attribute__((ext_vector_type(4)))  int i32x4;

namespace {
constexpr int kB  = 4;
constexpr int kL  = 2048;
constexpr int kD  = 512;
constexpr int kH  = 8;
constexpr int kDk = 64;
constexpr int kD2 = 1024;
constexpr int kTok = kB * kL;      // 8192 tokens
constexpr float kBnEps = 1e-5f;
constexpr float kScale = 0.125f;   // 1/sqrt(64)
}

#define DEVI static __device__ __forceinline__

DEVI v8f vzero8() {
  v8f z;
#pragma unroll
  for (int i = 0; i < 8; ++i) z[i] = 0.0f;
  return z;
}

DEVI v8f wmma_bf16(v16bf a, v16bf b, v8f c) {
  return __builtin_amdgcn_wmma_f32_16x16x32_bf16(false, a, false, b, (short)0, c,
                                                 false, false);
}

// A operand: 16x32 tile of a row-major [M x K] bf16 matrix, tile origin `base`.
// Lane l: row = l&15; k-values = {k8..k8+7, k8+16..k8+23}, k8 = 8*(l>>4).
DEVI v16bf load_fragA(const bf16* base, int ld) {
  const int lane = threadIdx.x & 31;
  const int row  = lane & 15;
  const int kb   = (lane >> 4) << 3;
  const bf16* p = base + (size_t)row * ld + kb;
  v8bf lo = *(const v8bf*)(p);
  v8bf hi = *(const v8bf*)(p + 16);
  v16bf f;
#pragma unroll
  for (int i = 0; i < 8; ++i) { f[i] = lo[i]; f[i + 8] = hi[i]; }
  return f;
}

// Same, but rows outside [rlo, rhi) supply zeros (conv boundary padding).
DEVI v16bf load_fragA_guard(const bf16* base, int ld, int rlo, int rhi) {
  const int lane = threadIdx.x & 31;
  const int row  = lane & 15;
  const int kb   = (lane >> 4) << 3;
  v16bf f;
  if (row >= rlo && row < rhi) {
    const bf16* p = base + (size_t)row * ld + kb;
    v8bf lo = *(const v8bf*)(p);
    v8bf hi = *(const v8bf*)(p + 16);
#pragma unroll
    for (int i = 0; i < 8; ++i) { f[i] = lo[i]; f[i + 8] = hi[i]; }
  } else {
#pragma unroll
    for (int i = 0; i < 16; ++i) f[i] = (bf16)0.0f;
  }
  return f;
}

// B operand: 32x16 tile of B, provided as row-major B^T [N x K], origin baseT.
// Lane l: n = l&15; k-values = {k8..k8+7, k8+16..k8+23}, k8 = 8*(l>>4).
DEVI v16bf load_fragB(const bf16* baseT, int ld) {
  const int lane = threadIdx.x & 31;
  const int n    = lane & 15;
  const int k0   = (lane >> 4) << 3;
  const bf16* p = baseT + (size_t)n * ld + k0;
  v8bf lo = *(const v8bf*)(p);
  v8bf hi = *(const v8bf*)(p + 16);
  v16bf f;
#pragma unroll
  for (int i = 0; i < 8; ++i) { f[i] = lo[i]; f[i + 8] = hi[i]; }
  return f;
}

DEVI float hmax16(float v) {
  v = fmaxf(v, __shfl_xor(v, 1, 16));
  v = fmaxf(v, __shfl_xor(v, 2, 16));
  v = fmaxf(v, __shfl_xor(v, 4, 16));
  v = fmaxf(v, __shfl_xor(v, 8, 16));
  return v;
}
DEVI float hsum16(float v) {
  v += __shfl_xor(v, 1, 16);
  v += __shfl_xor(v, 2, 16);
  v += __shfl_xor(v, 4, 16);
  v += __shfl_xor(v, 8, 16);
  return v;
}

// Generic-pointer low 32 bits == LDS byte offset (ISA 10.2 aperture mapping).
DEVI unsigned lds_addr32(const void* p) { return (unsigned)(size_t)p; }

// Issue a 2-D TDM tile load (bf16 elements) Global -> LDS.
// tile_d0 contiguous elements per row, tile_d1 rows, row stride stride0 elems.
// D# per ISA 8.3/8.4: group0 {count|lds|gaddr|type=2}, group1 {dsize|dims|tiles|strides}.
DEVI void tdm_load_2d(const void* gptr, unsigned ldsoff,
                      unsigned tensor_d0, unsigned tensor_d1,
                      unsigned long long stride0,
                      unsigned tile_d0, unsigned tile_d1) {
  const unsigned long long ga = (unsigned long long)(size_t)gptr;
  u32x4 g0;
  g0[0] = 1u;                                            // count=1 (valid, user)
  g0[1] = ldsoff;                                        // lds_addr
  g0[2] = (unsigned)(ga & 0xffffffffu);                  // global_addr[31:0]
  g0[3] = (unsigned)((ga >> 32) & 0x01ffffffu) | (2u << 30);  // [56:32] | type=2
  i32x8 g1;
  g1[0] = (int)(1u << 16);                               // data_size=1 -> 2 bytes
  g1[1] = (int)((tensor_d0 & 0xffffu) << 16);            // tensor_dim0[15:0]
  g1[2] = (int)(((tensor_d0 >> 16) & 0xffffu) |
                ((tensor_d1 & 0xffffu) << 16));          // dim0[31:16] | dim1[15:0]
  g1[3] = (int)(((tensor_d1 >> 16) & 0xffffu) |
                ((tile_d0 & 0xffffu) << 16));            // dim1[31:16] | tile_dim0
  g1[4] = (int)(tile_d1 & 0xffffu);                      // tile_dim1, tile_dim2=0
  g1[5] = (int)(stride0 & 0xffffffffull);                // dim0_stride[31:0]
  g1[6] = (int)((stride0 >> 32) & 0xffffull);            // dim0_stride[47:32]
  g1[7] = 0;
  i32x4 gz4 = {0, 0, 0, 0};
  i32x8 gz8 = {0, 0, 0, 0, 0, 0, 0, 0};
  // 6-arg toolchain variant: (g0, g1, g2, g3, extra_group, cpol)
  __builtin_amdgcn_tensor_load_to_lds(g0, g1, gz4, gz4, gz8, 0);
}

// ---------------------------------------------------------------- converts
__global__ void cvt_f32_bf16(const float* __restrict__ s, bf16* __restrict__ d, int n) {
  int i = blockIdx.x * blockDim.x + threadIdx.x;
  if (i < n) d[i] = (bf16)s[i];
}

// conv_w [Cout, Cin, taps] -> tap-major bf16 [taps][Cout][Cin]
__global__ void pack_conv_taps(const float* __restrict__ w, bf16* __restrict__ d, int taps) {
  int i = blockIdx.x * blockDim.x + threadIdx.x;
  int n = kD * kD * taps;
  if (i < n) {
    int t  = i % taps;
    int oi = i / taps;
    d[(size_t)t * kD * kD + oi] = (bf16)w[i];
  }
}

// ------------------------------------------------------- generic WMMA GEMM
// C[M,N] = A[M,K] * W[N,K]^T + bias.  K-loop unrolled 2x with alternating
// register sets (no rotation movs): set1 loads while set0 computes and v.v.
// mode 0: bf16 row-major out; mode 1: bf16 Vt[b,h,dk,L] out; mode 2: f32 out.
__global__ void __launch_bounds__(256)
gemm_wmma(const bf16* __restrict__ A, const bf16* __restrict__ W,
          const float* __restrict__ bias, bf16* __restrict__ obf,
          float* __restrict__ of32, int M, int N, int K, int mode) {
  const int lane = threadIdx.x & 31;
  const int wave = threadIdx.x >> 5;
  const int gw = blockIdx.x * 8 + wave;
  const int ntiles = N >> 6;
  const int mt = gw / ntiles;
  const int nt = gw - mt * ntiles;
  if (mt * 16 >= M) return;

  const bf16* abase = A + (size_t)mt * 16 * K;
  const bf16* wbase = W + (size_t)nt * 64 * K;
  v8f acc[4] = {vzero8(), vzero8(), vzero8(), vzero8()};

  v16bf af0 = load_fragA(abase, K);
  v16bf wf0[4];
#pragma unroll
  for (int j = 0; j < 4; ++j) wf0[j] = load_fragB(wbase + (size_t)j * 16 * K, K);

  for (int ks = 0; ks < K; ks += 64) {   // K is a multiple of 64
    v16bf af1 = load_fragA(abase + ks + 32, K);
    v16bf wf1[4];
#pragma unroll
    for (int j = 0; j < 4; ++j)
      wf1[j] = load_fragB(wbase + (size_t)j * 16 * K + ks + 32, K);
#pragma unroll
    for (int j = 0; j < 4; ++j) acc[j] = wmma_bf16(af0, wf0[j], acc[j]);
    if (ks + 64 < K) {
      __builtin_prefetch(abase + ks + 128, 0, 1);
      af0 = load_fragA(abase + ks + 64, K);
#pragma unroll
      for (int j = 0; j < 4; ++j)
        wf0[j] = load_fragB(wbase + (size_t)j * 16 * K + ks + 64, K);
    }
#pragma unroll
    for (int j = 0; j < 4; ++j) acc[j] = wmma_bf16(af1, wf1[j], acc[j]);
  }

  const int mloc = (lane >> 4) << 3;
  const int nloc = lane & 15;
#pragma unroll
  for (int j = 0; j < 4; ++j) {
    const int n = nt * 64 + j * 16 + nloc;
    const float bv = bias[n];
#pragma unroll
    for (int r = 0; r < 8; ++r) {
      const int m = mt * 16 + mloc + r;
      const float v = acc[j][r] + bv;
      if (mode == 0) {
        obf[(size_t)m * N + n] = (bf16)v;
      } else if (mode == 1) {
        const int b = m / kL, l = m - b * kL;
        const int h = n >> 6, dd = n & 63;
        obf[((size_t)(b * kH + h) * kDk + dd) * kL + l] = (bf16)v;
      } else {
        of32[(size_t)m * N + n] = v;
      }
    }
  }
}

// ------------------------------------------- multi-scale conv1d + BN (inference)
__global__ void __launch_bounds__(256)
conv_bn_wmma(const bf16* __restrict__ kin, const bf16* __restrict__ w3t,
             const bf16* __restrict__ w5t, const float* __restrict__ cb3,
             const float* __restrict__ cb5, const float* __restrict__ gamma,
             const float* __restrict__ betap, const float* __restrict__ mean,
             const float* __restrict__ var, bf16* __restrict__ klocal) {
  const int lane = threadIdx.x & 31;
  const int wave = threadIdx.x >> 5;
  const int gw = blockIdx.x * 8 + wave;   // 0..8191
  const int cht = gw & 15;
  const int lt  = (gw >> 4) & 127;
  const int b   = gw >> 11;
  const int ch0 = cht << 6;
  const bool is3 = ch0 < kD;
  const int taps = is3 ? 3 : 5;
  const int pad  = is3 ? 1 : 2;
  const int och0 = is3 ? ch0 : (ch0 - kD);
  const bf16* wt = is3 ? w3t : w5t;
  const float* cb = is3 ? cb3 : cb5;
  const int l0 = lt << 4;

  v8f acc[4] = {vzero8(), vzero8(), vzero8(), vzero8()};
  for (int t = 0; t < taps; ++t) {
    const int off = t - pad;
    int rlo = 0, rhi = 16;
    if (l0 + off < 0) rlo = -(l0 + off);
    if (l0 + off + 16 > kL) rhi = kL - (l0 + off);
    const bf16* abase = kin + ((size_t)b * kL + (size_t)(l0 + off)) * kD;
    const bf16* wb = wt + (size_t)t * kD * kD + (size_t)och0 * kD;

    v16bf af0 = load_fragA_guard(abase, kD, rlo, rhi);
    v16bf wf0[4];
#pragma unroll
    for (int j = 0; j < 4; ++j) wf0[j] = load_fragB(wb + (size_t)j * 16 * kD, kD);

    for (int ks = 0; ks < kD; ks += 64) {   // kD is a multiple of 64
      v16bf af1 = load_fragA_guard(abase + ks + 32, kD, rlo, rhi);
      v16bf wf1[4];
#pragma unroll
      for (int j = 0; j < 4; ++j)
        wf1[j] = load_fragB(wb + (size_t)j * 16 * kD + ks + 32, kD);
#pragma unroll
      for (int j = 0; j < 4; ++j) acc[j] = wmma_bf16(af0, wf0[j], acc[j]);
      if (ks + 64 < kD) {
        af0 = load_fragA_guard(abase + ks + 64, kD, rlo, rhi);
#pragma unroll
        for (int j = 0; j < 4; ++j)
          wf0[j] = load_fragB(wb + (size_t)j * 16 * kD + ks + 64, kD);
      }
#pragma unroll
      for (int j = 0; j < 4; ++j) acc[j] = wmma_bf16(af1, wf1[j], acc[j]);
    }
  }
  const int mloc = (lane >> 4) << 3;
  const int nloc = lane & 15;
#pragma unroll
  for (int j = 0; j < 4; ++j) {
    const int ch = ch0 + j * 16 + nloc;
    const int oc = och0 + j * 16 + nloc;
    const float s  = gamma[ch] * rsqrtf(var[ch] + kBnEps);
    const float sh = betap[ch] + (cb[oc] - mean[ch]) * s;
#pragma unroll
    for (int r = 0; r < 8; ++r) {
      const int l = l0 + mloc + r;
      klocal[((size_t)b * kL + l) * kD2 + ch] = (bf16)(acc[j][r] * s + sh);
    }
  }
}

// ------------------------------------------------- flash attention (one path)
// Block = one (b,h) x 128 Q rows (8 waves x 16-row tiles). K/V 32-key tiles are
// DMA'd by the Tensor Data Mover into double-buffered LDS shared by all waves;
// wave 0 issues TDM + s_wait_tensorcnt, block barriers publish the tiles.
__global__ void __launch_bounds__(256)
attn_wmma(const bf16* __restrict__ qm, const bf16* __restrict__ km,
          const bf16* __restrict__ vt, float* __restrict__ x, int accumulate) {
  __shared__ __align__(16) bf16 kbuf[2][32 * 64];   // [key][dk]   4KB each
  __shared__ __align__(16) bf16 vbuf[2][64 * 32];   // [dk][key]   4KB each
  __shared__ __align__(16) bf16 plds[8][16 * 32];   // per-wave P tile
  const int lane = threadIdx.x & 31;
  const int wave = threadIdx.x >> 5;
  const int bh = blockIdx.x >> 4;          // 0..31
  const int qg = blockIdx.x & 15;          // 16 q-groups of 128 rows
  const int b = bh >> 3, h = bh & 7;
  const int qt = qg * 8 + wave;            // this wave's 16-row q tile

  const bf16* qbase = qm + ((size_t)b * kL + (size_t)qt * 16) * kD + h * 64;
  const bf16* kbs   = km + (size_t)b * kL * kD + h * 64;   // keys, ld=kD
  const bf16* vtb   = vt + (size_t)bh * kDk * kL;          // Vt rows=dk, ld=kL

  const v16bf qf0 = load_fragA(qbase, kD);
  const v16bf qf1 = load_fragA(qbase + 32, kD);

  v8f acc[4] = {vzero8(), vzero8(), vzero8(), vzero8()};
  float rmax[8], rsum[8];
#pragma unroll
  for (int r = 0; r < 8; ++r) { rmax[r] = -3.0e38f; rsum[r] = 0.0f; }

  bf16* pl = &plds[wave][0];
  const int mrow = (lane >> 4) << 3;
  const int nloc = lane & 15;

  if (wave == 0) {  // prologue: DMA tile 0
    tdm_load_2d(kbs, lds_addr32(&kbuf[0][0]), kD, 32, kD, 64, 32);
    tdm_load_2d(vtb, lds_addr32(&vbuf[0][0]), kL, 64, kL, 32, 64);
  }

  for (int kt = 0; kt < kL; kt += 32) {
    const int buf = (kt >> 5) & 1;
    if (wave == 0) {
      if (kt + 32 < kL) {   // DMA next tile into the other buffer
        tdm_load_2d(kbs + (size_t)(kt + 32) * kD, lds_addr32(&kbuf[buf ^ 1][0]),
                    kD, 32, kD, 64, 32);
        tdm_load_2d(vtb + (kt + 32), lds_addr32(&vbuf[buf ^ 1][0]),
                    kL, 64, kL, 32, 64);
        __builtin_amdgcn_s_wait_tensorcnt(2);   // current tile's 2 DMAs done
      } else {
        __builtin_amdgcn_s_wait_tensorcnt(0);
      }
    }
    __syncthreads();                            // publish current tile
    const bf16* kl_ = &kbuf[buf][0];
    const bf16* vl_ = &vbuf[buf][0];

    // S tiles: 16 q-rows x 32 keys, K=64 over two k-steps (operands from LDS)
    v8f s0 = vzero8(), s1 = vzero8();
    {
      v16bf kf = load_fragB(kl_, 64);
      s0 = wmma_bf16(qf0, kf, s0);
      kf = load_fragB(kl_ + 32, 64);
      s0 = wmma_bf16(qf1, kf, s0);
      kf = load_fragB(kl_ + 16 * 64, 64);
      s1 = wmma_bf16(qf0, kf, s1);
      kf = load_fragB(kl_ + 16 * 64 + 32, 64);
      s1 = wmma_bf16(qf1, kf, s1);
    }
    // online softmax per row (rows live in 16-lane halves, one per C-VGPR)
#pragma unroll
    for (int r = 0; r < 8; ++r) {
      const float a0 = s0[r] * kScale;
      const float a1 = s1[r] * kScale;
      float mx = fmaxf(hmax16(a0), hmax16(a1));
      mx = fmaxf(mx, rmax[r]);
      const float alpha = __expf(rmax[r] - mx);
      rmax[r] = mx;
      const float p0 = __expf(a0 - mx);
      const float p1 = __expf(a1 - mx);
      rsum[r] = rsum[r] * alpha + hsum16(p0) + hsum16(p1);
      acc[0][r] = acc[0][r] * alpha;
      acc[1][r] = acc[1][r] * alpha;
      acc[2][r] = acc[2][r] * alpha;
      acc[3][r] = acc[3][r] * alpha;
      pl[(mrow + r) * 32 + nloc]      = (bf16)p0;
      pl[(mrow + r) * 32 + 16 + nloc] = (bf16)p1;
    }
    // order per-wave LDS stores before the A-fragment re-gather
    asm volatile("s_wait_dscnt 0x0" ::: "memory");
    v16bf pf = load_fragA(pl, 32);
#pragma unroll
    for (int j = 0; j < 4; ++j) {
      v16bf vf = load_fragB(vl_ + (size_t)j * 16 * 32, 32);
      acc[j] = wmma_bf16(pf, vf, acc[j]);
    }
    __syncthreads();   // all waves done with `buf` before TDM reuses it
  }

  float* xb = x + ((size_t)b * kL + (size_t)qt * 16) * kD + h * 64;
#pragma unroll
  for (int r = 0; r < 8; ++r) {
    const float inv = 1.0f / rsum[r];
    const size_t o = (size_t)(mrow + r) * kD + nloc;
    if (accumulate) {
      xb[o]      += acc[0][r] * inv;
      xb[o + 16] += acc[1][r] * inv;
      xb[o + 32] += acc[2][r] * inv;
      xb[o + 48] += acc[3][r] * inv;
    } else {
      xb[o]      = acc[0][r] * inv;
      xb[o + 16] = acc[1][r] * inv;
      xb[o + 32] = acc[2][r] * inv;
      xb[o + 48] = acc[3][r] * inv;
    }
  }
}

// ---------------------------------------------------------------- host side
extern "C" void kernel_launch(void* const* d_in, const int* in_sizes, int n_in,
                              void* d_out, int out_size, void* d_ws, size_t ws_size,
                              hipStream_t stream) {
  const float* query   = (const float*)d_in[0];
  const float* key_    = (const float*)d_in[1];
  const float* value   = (const float*)d_in[2];
  const float* conv_w3 = (const float*)d_in[3];
  const float* conv_b3 = (const float*)d_in[4];
  const float* conv_w5 = (const float*)d_in[5];
  const float* conv_b5 = (const float*)d_in[6];
  const float* bn_gamma = (const float*)d_in[7];
  const float* bn_beta  = (const float*)d_in[8];
  const float* bn_mean  = (const float*)d_in[9];
  const float* bn_var   = (const float*)d_in[10];
  const float* wq  = (const float*)d_in[11];
  const float* bq  = (const float*)d_in[12];
  const float* wkl = (const float*)d_in[13];
  const float* bkl = (const float*)d_in[14];
  const float* wkg = (const float*)d_in[15];
  const float* bkg = (const float*)d_in[16];
  const float* wv  = (const float*)d_in[17];
  const float* bv  = (const float*)d_in[18];
  const float* wo  = (const float*)d_in[19];
  const float* bo  = (const float*)d_in[20];
  (void)in_sizes; (void)n_in; (void)out_size; (void)ws_size;

  char* ws = (char*)d_ws;
  size_t off = 0;
  auto take = [&](size_t bytes) -> void* {
    void* p = ws + off;
    off = (off + bytes + 255) & ~(size_t)255;
    return p;
  };
  const size_t tokD  = (size_t)kTok * kD;
  const size_t tokD2 = (size_t)kTok * kD2;

  bf16* qin   = (bf16*)take(tokD * 2);
  bf16* kin   = (bf16*)take(tokD * 2);
  bf16* vin   = (bf16*)take(tokD * 2);
  bf16* wq_b  = (bf16*)take((size_t)kD * kD * 2);
  bf16* wkl_b = (bf16*)take((size_t)kD * kD2 * 2);
  bf16* wkg_b = (bf16*)take((size_t)kD * kD * 2);
  bf16* wv_b  = (bf16*)take((size_t)kD * kD * 2);
  bf16* wo_b  = (bf16*)take((size_t)kD * kD * 2);
  bf16* w3t   = (bf16*)take((size_t)3 * kD * kD * 2);
  bf16* w5t   = (bf16*)take((size_t)5 * kD * kD * 2);
  bf16* klocal= (bf16*)take(tokD2 * 2);
  bf16* qp    = (bf16*)take(tokD * 2);
  bf16* klp   = (bf16*)take(tokD * 2);
  bf16* kgp   = (bf16*)take(tokD * 2);
  bf16* vt    = (bf16*)take(tokD * 2);
  float* xf   = (float*)take(tokD * 4);
  bf16* xb    = (bf16*)take(tokD * 2);

  const int T = 256;
  cvt_f32_bf16<<<(int)((tokD + T - 1) / T), T, 0, stream>>>(query, qin, (int)tokD);
  cvt_f32_bf16<<<(int)((tokD + T - 1) / T), T, 0, stream>>>(key_, kin, (int)tokD);
  cvt_f32_bf16<<<(int)((tokD + T - 1) / T), T, 0, stream>>>(value, vin, (int)tokD);
  const int nw = kD * kD;
  cvt_f32_bf16<<<(nw + T - 1) / T, T, 0, stream>>>(wq, wq_b, nw);
  cvt_f32_bf16<<<(kD * kD2 + T - 1) / T, T, 0, stream>>>(wkl, wkl_b, kD * kD2);
  cvt_f32_bf16<<<(nw + T - 1) / T, T, 0, stream>>>(wkg, wkg_b, nw);
  cvt_f32_bf16<<<(nw + T - 1) / T, T, 0, stream>>>(wv, wv_b, nw);
  cvt_f32_bf16<<<(nw + T - 1) / T, T, 0, stream>>>(wo, wo_b, nw);
  pack_conv_taps<<<(3 * nw + T - 1) / T, T, 0, stream>>>(conv_w3, w3t, 3);
  pack_conv_taps<<<(5 * nw + T - 1) / T, T, 0, stream>>>(conv_w5, w5t, 5);

  // conv(k=3,5) + BN -> key_local bf16 [B,L,1024]
  conv_bn_wmma<<<1024, 256, 0, stream>>>(kin, w3t, w5t, conv_b3, conv_b5,
                                         bn_gamma, bn_beta, bn_mean, bn_var, klocal);

  // projections (8192x512 GEMMs)
  gemm_wmma<<<512, 256, 0, stream>>>(qin,    wq_b,  bq,  qp,  nullptr, kTok, kD, kD,  0);
  gemm_wmma<<<512, 256, 0, stream>>>(klocal, wkl_b, bkl, klp, nullptr, kTok, kD, kD2, 0);
  gemm_wmma<<<512, 256, 0, stream>>>(kin,    wkg_b, bkg, kgp, nullptr, kTok, kD, kD,  0);
  gemm_wmma<<<512, 256, 0, stream>>>(vin,    wv_b,  bv,  vt,  nullptr, kTok, kD, kD,  1);

  // dual-path flash attention: softmax(QKl^T)V then += softmax(QKg^T)V
  attn_wmma<<<512, 256, 0, stream>>>(qp, klp, vt, xf, 0);
  attn_wmma<<<512, 256, 0, stream>>>(qp, kgp, vt, xf, 1);

  cvt_f32_bf16<<<(int)((tokD + T - 1) / T), T, 0, stream>>>(xf, xb, (int)tokD);

  // output projection -> f32 d_out
  gemm_wmma<<<512, 256, 0, stream>>>(xb, wo_b, bo, nullptr, (float*)d_out,
                                     kTok, kD, kD, 2);
}
